// AFM_82248623718441
// MI455X (gfx1250) — matrix operand order repaired and checked by reading, not compiled, not run on hardware
//
#include <hip/hip_runtime.h>

// AFM forward for MI455X (gfx1250, wave32, WMMA).
// One workgroup per batch row. Attention GEMM via v_wmma_f32_16x16x32_f16.
// Score reduction uses a value-folding butterfly: 8 shuffles (not 32) per tile,
// one predicated LDS store.

typedef __attribute__((ext_vector_type(16))) _Float16 v16h;
typedef __attribute__((ext_vector_type(8)))  float    v8f;

#define NF     64          // n features
#define DE     64          // embedding dim
#define NATT   64          // attention dim
#define NPAIR  2016        // 64*63/2
#define NTILE  126         // NPAIR / 16
#define BLOCK  256         // 8 waves of 32
#define NWAVE  8

__global__ __launch_bounds__(BLOCK)
void afm_fwd_kernel(const float* __restrict__ x,    const float* __restrict__ V,
                    const float* __restrict__ w1,   const float* __restrict__ b1,
                    const float* __restrict__ w2,   const float* __restrict__ b2,
                    const float* __restrict__ wfc,  const float* __restrict__ bfc,
                    const float* __restrict__ wlin, const float* __restrict__ blin,
                    float* __restrict__ out)
{
    __shared__ float          eLds[NF * DE];    // e[i][d]   (bank-friendly for pooled pass)
    __shared__ float          eT  [DE * NF];    // e[d][i]   (bank-friendly for A-fragment build)
    __shared__ float          scoreLds[NPAIR];  // scores, then exp values
    __shared__ unsigned short pairLds[NPAIR];   // packed (i<<8)|j
    __shared__ float          red[BLOCK];       // reduction scratch

    const int b    = blockIdx.x;
    const int tid  = threadIdx.x;
    const int lane = tid & 31;
    const int wave = tid >> 5;
    const int nlo  = lane & 15;       // position within 16-lane half
    const int hi   = lane >> 4;       // 0: lanes 0-15, 1: lanes 16-31

    // ---- e = diag(x_b) * V : e[i][d] = x[b][i] * V[i][d] ----
    for (int idx = tid; idx < NF * DE; idx += BLOCK) {
        int i = idx >> 6, d = idx & 63;
        float v = x[b * NF + i] * V[idx];
        eLds[idx]      = v;
        eT[d * NF + i] = v;
    }
    // ---- triu pair table (row-major (0,1),(0,2),...,(62,63)) ----
    for (int p = tid; p < NPAIR; p += BLOCK) {
        int i = 0, base = 0;
        while (base + (NF - 1 - i) <= p) { base += NF - 1 - i; ++i; }
        int j = i + 1 + (p - base);
        pairLds[p] = (unsigned short)((i << 8) | j);
    }
    __syncthreads();

    // ---- loop-invariant per-lane fragments ----
    // B fragment (K x N) for  h = prod @ w1^T :  B[k][n] = w1[n][k]  (f32 -> f16)
    // lanes 0-15: K = c*32 + {0..7,16..23} ; lanes 16-31: +8 ; n = nt*16 + nlo
    v16h bfrag[4][2];
    #pragma unroll
    for (int nt = 0; nt < 4; ++nt) {
        const int n = nt * 16 + nlo;
        #pragma unroll
        for (int c = 0; c < 2; ++c) {
            const int kb = c * 32 + hi * 8;
            #pragma unroll
            for (int e = 0; e < 16; ++e)
                bfrag[nt][c][e] = (_Float16)w1[n * DE + kb + ((e < 8) ? e : e + 8)];
        }
    }
    float b1v[4], w2v[4];
    #pragma unroll
    for (int nt = 0; nt < 4; ++nt) {
        b1v[nt] = b1[nt * 16 + nlo];
        w2v[nt] = w2[nt * 16 + nlo];
    }

    // ---- pair-tile loop: 8 WMMAs per 16-pair tile, fused relu+w2 reduction ----
    // wave-uniform trip count per wave -> EXEC all-1s at every WMMA.
    const int akb = hi * 8;   // A-layout K base for hi lane-half
    for (int t = wave; t < NTILE; t += NWAVE) {
        const int p0 = t * 16;
        const int pk = pairLds[p0 + nlo];
        const int pi = pk >> 8, pj = pk & 255;
        const float* ei = &eT[pi];            // eT[d*64 + i]
        const float* ej = &eT[pj];
        // A fragments: prod[p, d], 16-bit A 16x32 layout:
        // lanes 0-15: K in {0..7,16..23}; lanes 16-31: +8
        v16h a0, a1;
        #pragma unroll
        for (int e = 0; e < 16; ++e) {
            const int K = ((e < 8) ? e : e + 8) + akb;
            a0[e] = (_Float16)(ei[K * NF]        * ej[K * NF]);
            a1[e] = (_Float16)(ei[(K + 32) * NF] * ej[(K + 32) * NF]);
        }
        // All 8 WMMAs into 4 independent accumulators first (scheduler freedom).
        v8f acc[4];
        #pragma unroll
        for (int nt = 0; nt < 4; ++nt) {
            v8f c0 = {};
            c0 = __builtin_amdgcn_wmma_f32_16x16x32_f16(false, a0, false, bfrag[nt][0],
                                                        (short)0, c0, false, false);
            acc[nt] = __builtin_amdgcn_wmma_f32_16x16x32_f16(false, a1, false, bfrag[nt][1],
                                                             (short)0, c0, false, false);
        }
        // s[r] = per-lane contribution to score of row M = r + 8*hi
        float s[8];
        #pragma unroll
        for (int r = 0; r < 8; ++r) {
            float acc_s = 0.f;
            #pragma unroll
            for (int nt = 0; nt < 4; ++nt) {
                float h = acc[nt][r] + b1v[nt];          // + b_att1
                acc_s += (h > 0.f ? h : 0.f) * w2v[nt];  // relu, * w_att2
            }
            s[r] = acc_s;
        }
        // Value-folding butterfly over the 16 N-lanes of each half: 8 shuffles.
        // After folding on masks 8/4/2, lane nlo tracks row r whose bits 2..0
        // are nlo bits 3..1; mask-1 stage completes the 16-lane sum.
        float t4[4];
        #pragma unroll
        for (int r = 0; r < 4; ++r) {                     // mask 8: fold 8 -> 4
            float mine  = (nlo & 8) ? s[r + 4] : s[r];
            float other = (nlo & 8) ? s[r]     : s[r + 4];
            t4[r] = mine + __shfl_xor(other, 8, 32);
        }
        float t2[2];
        #pragma unroll
        for (int q = 0; q < 2; ++q) {                     // mask 4: fold 4 -> 2
            float mine  = (nlo & 4) ? t4[q + 2] : t4[q];
            float other = (nlo & 4) ? t4[q]     : t4[q + 2];
            t2[q] = mine + __shfl_xor(other, 4, 32);
        }
        {
            float mine  = (nlo & 2) ? t2[1] : t2[0];      // mask 2: fold 2 -> 1
            float other = (nlo & 2) ? t2[0] : t2[1];
            float v = mine + __shfl_xor(other, 2, 32);
            v += __shfl_xor(v, 1, 32);                    // mask 1: plain reduce
            // even lane nlo holds score of row r = nlo>>1; single predicated store.
            if (!(nlo & 1)) scoreLds[p0 + hi * 8 + (nlo >> 1)] = v;
            // note: + b_att2 omitted — softmax is shift-invariant.
        }
    }
    __syncthreads();

    // ---- softmax over the 2016 pairs ----
    float lm = -1e30f;
    for (int p = tid; p < NPAIR; p += BLOCK) lm = fmaxf(lm, scoreLds[p]);
    red[tid] = lm; __syncthreads();
    for (int st = BLOCK / 2; st > 0; st >>= 1) {
        if (tid < st) red[tid] = fmaxf(red[tid], red[tid + st]);
        __syncthreads();
    }
    const float mx = red[0]; __syncthreads();

    float lsum = 0.f;
    for (int p = tid; p < NPAIR; p += BLOCK) {
        float ev = __expf(scoreLds[p] - mx);
        scoreLds[p] = ev;                  // keep unnormalized weights
        lsum += ev;
    }
    __syncthreads();
    red[tid] = lsum; __syncthreads();
    for (int st = BLOCK / 2; st > 0; st >>= 1) {
        if (tid < st) red[tid] += red[tid + st];
        __syncthreads();
    }
    const float denom = red[0]; __syncthreads();

    // ---- pooled[d] = (1/denom) * sum_p exp[p] * e[i_p][d] * e[j_p][d] ----
    const int d = tid & 63, g = tid >> 6;   // 4 partial accumulators per d
    float part = 0.f;
    for (int p = g; p < NPAIR; p += 4) {
        const int pk = pairLds[p];
        const int pi = pk >> 8, pj = pk & 255;
        part += scoreLds[p] * eLds[pi * DE + d] * eLds[pj * DE + d];
    }
    red[tid] = part; __syncthreads();
    if (tid < 64) {
        float pooled = (red[tid] + red[tid + 64] + red[tid + 128] + red[tid + 192]) / denom;
        red[tid] = pooled * wfc[tid] + x[b * NF + tid] * wlin[tid];
    }
    __syncthreads();
    if (tid == 0) {
        float logit = bfc[0] + blin[0];
        #pragma unroll
        for (int k = 0; k < 64; ++k) logit += red[k];
        out[b] = 1.f / (1.f + __expf(-logit));
    }
}

extern "C" void kernel_launch(void* const* d_in, const int* in_sizes, int n_in,
                              void* d_out, int out_size, void* d_ws, size_t ws_size,
                              hipStream_t stream) {
    const float* x    = (const float*)d_in[0];
    const float* V    = (const float*)d_in[1];
    const float* w1   = (const float*)d_in[2];
    const float* b1   = (const float*)d_in[3];
    const float* w2   = (const float*)d_in[4];
    const float* b2   = (const float*)d_in[5];
    const float* wfc  = (const float*)d_in[6];
    const float* bfc  = (const float*)d_in[7];
    const float* wlin = (const float*)d_in[8];
    const float* blin = (const float*)d_in[9];
    float* out = (float*)d_out;

    const int B = in_sizes[0] / NF;   // 2048
    afm_fwd_kernel<<<B, BLOCK, 0, stream>>>(x, V, w1, b1, w2, b2,
                                            wfc, bfc, wlin, blin, out);
    (void)d_ws; (void)ws_size; (void)n_in; (void)out_size;
    (void)b2;
}
